// SwinTransformerBlock_29076928594377
// MI455X (gfx1250) — matrix operand order repaired and checked by reading, not compiled, not run on hardware
//
#include <hip/hip_runtime.h>

typedef __bf16 bf16;
typedef __attribute__((ext_vector_type(16))) __bf16 v16bf;
typedef __attribute__((ext_vector_type(8)))  __bf16 v8bf;
typedef __attribute__((ext_vector_type(8)))  float   v8f;

#define WMMA_BF16(A, B, C) \
  __builtin_amdgcn_wmma_f32_16x16x32_bf16(false, (A), false, (B), (short)0, (C), false, false)

__device__ __forceinline__ void lds_fence() {
  asm volatile("s_wait_dscnt 0" ::: "memory");
}

__device__ __forceinline__ v8f splat8(float v) {
  v8f r;
#pragma unroll
  for (int i = 0; i < 8; ++i) r[i] = v;
  return r;
}

// A fragment (16x32 bf16) from row-major [.][ld]: lane lm = row, per-lane two
// contiguous 8-element chunks at k0+8*hi and k0+16+8*hi (ISA 7.12.2 layout).
__device__ __forceinline__ v16bf load_a_frag(const bf16* base, int ld, int row0,
                                             int k0, int lm, int hi) {
  const bf16* p = base + (size_t)(row0 + lm) * ld + k0;
  union { v16bf v; v8bf h[2]; } u;
  u.h[0] = *(const v8bf*)(p + 8 * hi);
  u.h[1] = *(const v8bf*)(p + 16 + 8 * hi);
  return u.v;
}

// B fragment (32x16 bf16) from column-major [n][k] storage with leading dim ldk:
// lane lm = column n, two 8-element chunks at k0+8*hi and k0+16+8*hi (matches A).
__device__ __forceinline__ v16bf load_b_frag(const bf16* base, int ldk, int n0,
                                             int k0, int lm, int hi) {
  const bf16* p = base + (size_t)(n0 + lm) * ldk + k0 + 8 * hi;
  union { v16bf v; v8bf h[2]; } u;
  u.h[0] = *(const v8bf*)(p);
  u.h[1] = *(const v8bf*)(p + 16);
  return u.v;
}

// dst[n*K + k] = bf16(src[k*N + n])  (weights -> [n][k] bf16)
__global__ void transpose_to_bf16(const float* __restrict__ src,
                                  bf16* __restrict__ dst, int K, int N) {
  int idx = blockIdx.x * blockDim.x + threadIdx.x;
  if (idx >= K * N) return;
  int n = idx / K;
  int k = idx - n * K;
  dst[idx] = (bf16)src[k * N + n];
}

// LDS region C layout (bytes), aliased across phases:
//   attention phase: q/o [0,13312) | k [13312,26624) | vT [26624,40448)
//                    | s [40448,57856) | p [57856,67072)
//   mlp phase:       act[64][392] bf16 = 50176 bytes at offset 0
#define REGC_BYTES 67072

__global__ __launch_bounds__(128) void swin_block_kernel(
    const float* __restrict__ x,
    const float* __restrict__ ln1g, const float* __restrict__ ln1b,
    const float* __restrict__ biasq, const float* __restrict__ biask,
    const float* __restrict__ biasv, const float* __restrict__ biaso,
    const float* __restrict__ ln2g, const float* __restrict__ ln2b,
    const float* __restrict__ bias1, const float* __restrict__ bias2,
    const bf16* __restrict__ wqkvT, const bf16* __restrict__ woT,
    const bf16* __restrict__ w1T, const bf16* __restrict__ w2T,
    float* __restrict__ out) {
  __shared__ float xwin[64 * 96];                      // 24576 B, f32 residual path
  __shared__ bf16 hbuf[64 * 104];                      // 13312 B, LN1/LN2 output
  __shared__ __align__(64) unsigned char regC[REGC_BYTES];

  bf16* qbuf = (bf16*)regC;                    // [64][104], also o-buffer later
  bf16* kbuf = (bf16*)(regC + 13312);          // [64][104]
  bf16* vT   = (bf16*)(regC + 26624);          // [96][72]  (d-major, token minor)
  float* sbuf = (float*)(regC + 40448);        // [4][16][68] per-wave scores
  bf16* pbuf = (bf16*)(regC + 57856);          // [4][16][72] per-wave probs
  bf16* act  = (bf16*)regC;                    // [64][392] MLP hidden (aliased)

  const int tid  = threadIdx.x;
  const int wave = tid >> 5;
  const int lane = tid & 31;
  const int lm   = lane & 15;
  const int hi   = lane >> 4;
  const int row0 = wave * 16;

  const int blk = blockIdx.x;
  const int b   = blk >> 10;
  const int rem = blk & 1023;
  const int wy  = rem >> 5;
  const int wx  = rem & 31;

  // ---- load shifted window: xs[i] = x[(i+4) mod 256] ----
  {
    int token = tid >> 1, part = tid & 1;
    int ty = token >> 3, tx = token & 7;
    int gy = (wy * 8 + ty + 4) & 255;
    int gx = (wx * 8 + tx + 4) & 255;
    const float4* src =
        (const float4*)(x + ((((size_t)b * 256 + gy) * 256 + gx) * 96) + part * 48);
    float4* dst = (float4*)(xwin + token * 96 + part * 48);
#pragma unroll
    for (int i = 0; i < 12; ++i) dst[i] = src[i];
  }
  lds_fence();

  // ---- LN1 (wave-local rows: lane pair (lm, lm+16) handles token row0+lm) ----
  {
    const int token = row0 + lm;
    const float* xr = xwin + token * 96 + hi * 48;
    float xl[48];
    float s = 0.f;
#pragma unroll
    for (int j = 0; j < 48; ++j) { xl[j] = xr[j]; s += xl[j]; }
    s += __shfl_xor(s, 16, 32);
    float mean = s * (1.0f / 96.0f);
    float vs = 0.f;
#pragma unroll
    for (int j = 0; j < 48; ++j) { float d = xl[j] - mean; vs += d * d; }
    vs += __shfl_xor(vs, 16, 32);
    float rstd = rsqrtf(vs * (1.0f / 96.0f) + 1e-6f);
    bf16* hr = hbuf + token * 104 + hi * 48;
#pragma unroll
    for (int j = 0; j < 48; ++j) {
      int c = hi * 48 + j;
      hr[j] = (bf16)((xl[j] - mean) * rstd * ln1g[c] + ln1b[c]);
    }
  }
  __syncthreads();  // hbuf + xwin published to all waves

  // ---- QKV: [64x96] x [96x288], column-partitioned over waves so each
  //      weight B-fragment is loaded from global exactly once per block and
  //      reused across all 4 M-tiles. Q scaled by 1/sqrt(32); V transposed.
  {
#pragma unroll 1
    for (int nn = wave; nn < 18; nn += 4) {
      int n0 = nn * 16;
      // tile type is wave-uniform; force a scalar branch (no EXEC masking)
      int n0u = __builtin_amdgcn_readfirstlane(n0);
      const float* bp = (n0u < 96) ? (biasq + n0u)
                      : (n0u < 192) ? (biask + (n0u - 96)) : (biasv + (n0u - 192));
      float bias = bp[lm];
      v16bf b0 = load_b_frag(wqkvT, 96, n0, 0, lm, hi);
      v16bf b1 = load_b_frag(wqkvT, 96, n0, 32, lm, hi);
      v16bf b2 = load_b_frag(wqkvT, 96, n0, 64, lm, hi);
#pragma unroll
      for (int m = 0; m < 4; ++m) {
        int r0 = m * 16;
        v8f acc = splat8(bias);
        acc = WMMA_BF16(load_a_frag(hbuf, 104, r0, 0, lm, hi), b0, acc);
        acc = WMMA_BF16(load_a_frag(hbuf, 104, r0, 32, lm, hi), b1, acc);
        acc = WMMA_BF16(load_a_frag(hbuf, 104, r0, 64, lm, hi), b2, acc);
        if (n0u < 96) {
          int col = n0 + lm;
#pragma unroll
          for (int r = 0; r < 8; ++r)
            qbuf[(r0 + r + 8 * hi) * 104 + col] =
                (bf16)(acc[r] * 0.17677669529663687f);
        } else if (n0u < 192) {
          int c2 = n0 + lm - 96;
#pragma unroll
          for (int r = 0; r < 8; ++r)
            kbuf[(r0 + r + 8 * hi) * 104 + c2] = (bf16)acc[r];
        } else {
          int c2 = n0 + lm - 192;
          // contiguous in r -> vectorizes to ds_store_b128 per lane
          bf16* vp = vT + c2 * 72 + r0 + 8 * hi;
#pragma unroll
          for (int r = 0; r < 8; ++r) vp[r] = (bf16)acc[r];
        }
      }
    }
  }
  __syncthreads();  // publish Q/K/vT across waves

  // ---- per-head attention (row-partitioned, wave-local; B operands in LDS) ----
  float* s_w = sbuf + wave * 16 * 68;
  bf16* p_w  = pbuf + wave * 16 * 72;
#pragma unroll 1
  for (int h = 0; h < 3; ++h) {
    v16bf aq = load_a_frag(qbuf, 104, row0, h * 32, lm, hi);
#pragma unroll
    for (int nn = 0; nn < 4; ++nn) {
      v8f acc = splat8(0.f);
      acc = WMMA_BF16(aq, load_b_frag(kbuf, 104, nn * 16, h * 32, lm, hi), acc);
#pragma unroll
      for (int r = 0; r < 8; ++r)
        s_w[(r + 8 * hi) * 68 + nn * 16 + lm] = acc[r];
    }
    lds_fence();
    {  // softmax: lane handles row lm, columns hi*32..hi*32+31
      float ex[32];
      const float* sr = s_w + lm * 68 + hi * 32;
      float m = -3.4e38f;
#pragma unroll
      for (int j = 0; j < 32; ++j) { ex[j] = sr[j]; m = fmaxf(m, ex[j]); }
      m = fmaxf(m, __shfl_xor(m, 16, 32));
      float sum = 0.f;
#pragma unroll
      for (int j = 0; j < 32; ++j) { ex[j] = __expf(ex[j] - m); sum += ex[j]; }
      sum += __shfl_xor(sum, 16, 32);
      float inv = 1.0f / sum;
      bf16* pr = p_w + lm * 72 + hi * 32;
#pragma unroll
      for (int j = 0; j < 32; ++j) pr[j] = (bf16)(ex[j] * inv);
    }
    lds_fence();
    {  // O_h = P [16x64] x V_h [64x32]; into o-buffer rows (aliases dead Q cols)
      v16bf ap0 = load_a_frag(p_w, 72, 0, 0, lm, hi);
      v16bf ap1 = load_a_frag(p_w, 72, 0, 32, lm, hi);
#pragma unroll
      for (int dn = 0; dn < 2; ++dn) {
        v8f acc = splat8(0.f);
        acc = WMMA_BF16(ap0, load_b_frag(vT, 72, h * 32 + dn * 16, 0, lm, hi), acc);
        acc = WMMA_BF16(ap1, load_b_frag(vT, 72, h * 32 + dn * 16, 32, lm, hi), acc);
        int c = h * 32 + dn * 16 + lm;
#pragma unroll
        for (int r = 0; r < 8; ++r)
          qbuf[(row0 + r + 8 * hi) * 104 + c] = (bf16)acc[r];  // o-buffer
      }
    }
    lds_fence();
  }
  __syncthreads();  // o-buffer published across waves

  // ---- output projection + residual into xwin (column-partitioned) ----
  {
#pragma unroll 1
    for (int nn = wave; nn < 6; nn += 4) {
      int n0 = nn * 16, c = n0 + lm;
      v16bf b0 = load_b_frag(woT, 96, n0, 0, lm, hi);
      v16bf b1 = load_b_frag(woT, 96, n0, 32, lm, hi);
      v16bf b2 = load_b_frag(woT, 96, n0, 64, lm, hi);
#pragma unroll
      for (int m = 0; m < 4; ++m) {
        int r0 = m * 16;
        v8f acc = splat8(biaso[c]);
        acc = WMMA_BF16(load_a_frag(qbuf, 104, r0, 0, lm, hi), b0, acc);
        acc = WMMA_BF16(load_a_frag(qbuf, 104, r0, 32, lm, hi), b1, acc);
        acc = WMMA_BF16(load_a_frag(qbuf, 104, r0, 64, lm, hi), b2, acc);
#pragma unroll
        for (int r = 0; r < 8; ++r)
          xwin[(r0 + r + 8 * hi) * 96 + c] += acc[r];
      }
    }
  }
  __syncthreads();  // xwin columns merged across waves

  // ---- LN2 (wave-local rows, from updated xwin) ----
  {
    const int token = row0 + lm;
    const float* xr = xwin + token * 96 + hi * 48;
    float xl[48];
    float s = 0.f;
#pragma unroll
    for (int j = 0; j < 48; ++j) { xl[j] = xr[j]; s += xl[j]; }
    s += __shfl_xor(s, 16, 32);
    float mean = s * (1.0f / 96.0f);
    float vs = 0.f;
#pragma unroll
    for (int j = 0; j < 48; ++j) { float d = xl[j] - mean; vs += d * d; }
    vs += __shfl_xor(vs, 16, 32);
    float rstd = rsqrtf(vs * (1.0f / 96.0f) + 1e-6f);
    bf16* hr = hbuf + token * 104 + hi * 48;
#pragma unroll
    for (int j = 0; j < 48; ++j) {
      int c = hi * 48 + j;
      hr[j] = (bf16)((xl[j] - mean) * rstd * ln2g[c] + ln2b[c]);
    }
  }
  __syncthreads();  // hbuf published; attention LDS region now dead -> act

  // ---- MLP1: [64x96] x [96x384] + b1, exact GELU (column-partitioned) ----
  {
#pragma unroll 1
    for (int nn = wave; nn < 24; nn += 4) {
      int n0 = nn * 16, c = n0 + lm;
      v16bf b0 = load_b_frag(w1T, 96, n0, 0, lm, hi);
      v16bf b1 = load_b_frag(w1T, 96, n0, 32, lm, hi);
      v16bf b2 = load_b_frag(w1T, 96, n0, 64, lm, hi);
#pragma unroll
      for (int m = 0; m < 4; ++m) {
        int r0 = m * 16;
        v8f acc = splat8(bias1[c]);
        acc = WMMA_BF16(load_a_frag(hbuf, 104, r0, 0, lm, hi), b0, acc);
        acc = WMMA_BF16(load_a_frag(hbuf, 104, r0, 32, lm, hi), b1, acc);
        acc = WMMA_BF16(load_a_frag(hbuf, 104, r0, 64, lm, hi), b2, acc);
#pragma unroll
        for (int r = 0; r < 8; ++r) {
          float v = acc[r];
          float g = 0.5f * v * (1.0f + erff(v * 0.70710678118654752f));
          act[(r0 + r + 8 * hi) * 392 + c] = (bf16)g;
        }
      }
    }
  }
  __syncthreads();  // act published across waves

  // ---- MLP2: [64x384] x [384x96] + b2, residual (column-partitioned) ----
  {
#pragma unroll 1
    for (int nn = wave; nn < 6; nn += 4) {
      int c = nn * 16 + lm;
      v8f accs[4];
#pragma unroll
      for (int m = 0; m < 4; ++m) accs[m] = splat8(bias2[c]);
#pragma unroll 1
      for (int kk = 0; kk < 12; ++kk) {
        v16bf bfr = load_b_frag(w2T, 384, nn * 16, kk * 32, lm, hi);
#pragma unroll
        for (int m = 0; m < 4; ++m)
          accs[m] = WMMA_BF16(load_a_frag(act, 392, m * 16, kk * 32, lm, hi),
                              bfr, accs[m]);
      }
#pragma unroll
      for (int m = 0; m < 4; ++m)
#pragma unroll
        for (int r = 0; r < 8; ++r)
          xwin[(m * 16 + r + 8 * hi) * 96 + c] += accs[m][r];
    }
  }
  __syncthreads();  // final xwin merged across waves

  // ---- store (window-reverse + roll(+4) == same index as load) ----
  {
    int token = tid >> 1, part = tid & 1;
    int ty = token >> 3, tx = token & 7;
    int gy = (wy * 8 + ty + 4) & 255;
    int gx = (wx * 8 + tx + 4) & 255;
    float4* dst =
        (float4*)(out + ((((size_t)b * 256 + gy) * 256 + gx) * 96) + part * 48);
    const float4* src = (const float4*)(xwin + token * 96 + part * 48);
#pragma unroll
    for (int i = 0; i < 12; ++i) dst[i] = src[i];
  }
}

extern "C" void kernel_launch(void* const* d_in, const int* in_sizes, int n_in,
                              void* d_out, int out_size, void* d_ws,
                              size_t ws_size, hipStream_t stream) {
  const float* x    = (const float*)d_in[0];
  const float* ln1g = (const float*)d_in[1];
  const float* ln1b = (const float*)d_in[2];
  const float* wq   = (const float*)d_in[3];
  const float* bq   = (const float*)d_in[4];
  const float* wk   = (const float*)d_in[5];
  const float* bk   = (const float*)d_in[6];
  const float* wv   = (const float*)d_in[7];
  const float* bv   = (const float*)d_in[8];
  const float* wo   = (const float*)d_in[9];
  const float* bo   = (const float*)d_in[10];
  const float* ln2g = (const float*)d_in[11];
  const float* ln2b = (const float*)d_in[12];
  const float* w1   = (const float*)d_in[13];
  const float* b1   = (const float*)d_in[14];
  const float* w2   = (const float*)d_in[15];
  const float* b2   = (const float*)d_in[16];

  // workspace: bf16 weights in [n][k] layout
  bf16* wqkvT = (bf16*)d_ws;                           // 288*96
  bf16* woT   = (bf16*)((char*)d_ws + 55296);          // 96*96
  bf16* w1T   = (bf16*)((char*)d_ws + 73728);          // 384*96
  bf16* w2T   = (bf16*)((char*)d_ws + 147456);         // 96*384

  auto launchT = [&](const float* src, bf16* dst, int K, int N) {
    int n = K * N;
    transpose_to_bf16<<<(n + 255) / 256, 256, 0, stream>>>(src, dst, K, N);
  };
  launchT(wq, wqkvT, 96, 96);
  launchT(wk, wqkvT + 96 * 96, 96, 96);
  launchT(wv, wqkvT + 2 * 96 * 96, 96, 96);
  launchT(wo, woT, 96, 96);
  launchT(w1, w1T, 96, 384);
  launchT(w2, w2T, 384, 96);

  int batches = in_sizes[0] / (256 * 256 * 96);
  dim3 grid(batches * 32 * 32);
  swin_block_kernel<<<grid, 128, 0, stream>>>(
      x, ln1g, ln1b, bq, bk, bv, bo, ln2g, ln2b, b1, b2, wqkvT, woT, w1T, w2T,
      (float*)d_out);
}
